// RNNDecoder_79147657331350
// MI455X (gfx1250) — compile-verified, matrix operand order checked by reading, not run on hardware
//
#include <hip/hip_runtime.h>
#include <hip/hip_bf16.h>

// ---------------------------------------------------------------------------
// Problem dimensions (fixed by the reference)
// ---------------------------------------------------------------------------
constexpr int kB  = 128;   // batch
constexpr int kS  = 256;   // encoder sequence length
constexpr int kHE = 512;   // encoder hidden
constexpr int kHD = 512;   // decoder hidden
constexpr int kE  = 256;   // embedding dim
constexpr int kC  = 256;   // vocab classes
constexpr int kT  = 32;    // decode steps
constexpr int kHA = 512;   // attention hidden

typedef unsigned short u16;

// WMMA vector types (gfx1250 wave32)
typedef __attribute__((ext_vector_type(16))) __bf16 v16bf;
typedef __attribute__((ext_vector_type(8)))  float  v8f;

__device__ inline u16 f2b(float f) {
    unsigned u = __float_as_uint(f);
    u += 0x7fffu + ((u >> 16) & 1u);        // round to nearest even
    return (u16)(u >> 16);
}
__device__ inline float b2f(u16 h) {
    return __uint_as_float(((unsigned)h) << 16);
}

// ---------------------------------------------------------------------------
// fp32 -> bf16 conversion (same layout)
// ---------------------------------------------------------------------------
__global__ void cvt_f2b_kernel(const float* __restrict__ src,
                               u16* __restrict__ dst, int n) {
    int i = blockIdx.x * blockDim.x + threadIdx.x;
    if (i < n) dst[i] = f2b(src[i]);
}

// fp32 [K,N] -> bf16 [N,K] (transpose while converting; one-time, small)
__global__ void cvt_f2b_transpose_kernel(const float* __restrict__ src,
                                         u16* __restrict__ dst, int K, int N) {
    int i = blockIdx.x * blockDim.x + threadIdx.x;
    if (i >= K * N) return;
    int n = i / K, k = i % K;
    dst[(size_t)n * K + k] = f2b(src[(size_t)k * N + n]);
}

// ---------------------------------------------------------------------------
// Fragment load: 16-bit 16x32 A/B fragment per ISA layout.
// `p` already points at (row, k0 + half*8); the fragment is two b128 loads
// at byte offsets 0 and 32. 16B alignment holds for all lds/ldw used here.
// ---------------------------------------------------------------------------
__device__ inline v16bf ldfrag(const u16* __restrict__ p) {
    union { uint4 q[2]; v16bf v; } u;
    u.q[0] = *(const uint4*)(p);
    u.q[1] = *(const uint4*)(p + 16);
    return u.v;
}

// ---------------------------------------------------------------------------
// WMMA GEMM:  Y[M,N] = act( A[M,K](bf16) * W[N,K](bf16)^T + bias )
//   One wave computes a 16x64 tile: 1 A fragment x 4 B fragments,
//   4 v_wmma per K-step, software-pipelined (next fragments preloaded
//   before current WMMAs so VMEM latency overlaps the matrix pipe).
//   ACT: 0 none, 1 tanh.  OUTB: true -> bf16 out (Cb), false -> fp32 (Cout).
// ---------------------------------------------------------------------------
template <int ACT, bool OUTB>
__global__ void gemm_wmma(const u16* __restrict__ A, int lda,
                          const u16* __restrict__ W, int ldw,
                          const float* __restrict__ bias,
                          float* __restrict__ Cout, int ldc,
                          u16* __restrict__ Cb, int ldcb,
                          int M, int N, int K) {
    int w = blockIdx.x * (blockDim.x >> 5) + (threadIdx.x >> 5);
    int tilesM = M >> 4;
    int tilesN = N >> 6;                         // 64-wide N tiles
    if (w >= tilesM * tilesN) return;            // wave-uniform exit
    int tm = w % tilesM, tn = w / tilesM;
    int row0 = tm << 4, col0 = tn << 6;

    int lane = threadIdx.x & 31;
    int half = lane >> 4;                        // K-half selector
    int lrow = lane & 15;                        // M (A) / N (B) lane index

    // Per-lane fragment pointers (advance 32 elements per K-step).
    const u16* pa  = A + (size_t)(row0 + lrow) * lda + (half << 3);
    const u16* pb0 = W + (size_t)(col0 + lrow) * ldw + (half << 3);
    const u16* pb1 = pb0 + (size_t)16 * ldw;
    const u16* pb2 = pb0 + (size_t)32 * ldw;
    const u16* pb3 = pb0 + (size_t)48 * ldw;

    v8f acc0 = {0.f, 0.f, 0.f, 0.f, 0.f, 0.f, 0.f, 0.f};
    v8f acc1 = acc0, acc2 = acc0, acc3 = acc0;

    // Prologue: load K-step 0 fragments.
    v16bf a  = ldfrag(pa);
    v16bf b0 = ldfrag(pb0);
    v16bf b1 = ldfrag(pb1);
    v16bf b2 = ldfrag(pb2);
    v16bf b3 = ldfrag(pb3);

    for (int k0 = 0;;) {
        bool more = (k0 + 32) < K;
        v16bf an, bn0, bn1, bn2, bn3;
        if (more) {
            // gfx1250 global_prefetch_b8: pull the K-step after next into L2/L0
            __builtin_prefetch(pa + 64, 0, 3);
            __builtin_prefetch(pb0 + 64, 0, 3);
            an  = ldfrag(pa + 32);
            bn0 = ldfrag(pb0 + 32);
            bn1 = ldfrag(pb1 + 32);
            bn2 = ldfrag(pb2 + 32);
            bn3 = ldfrag(pb3 + 32);
        }
        acc0 = __builtin_amdgcn_wmma_f32_16x16x32_bf16(
            false, a, false, b0, (short)0, acc0, false, false);
        acc1 = __builtin_amdgcn_wmma_f32_16x16x32_bf16(
            false, a, false, b1, (short)0, acc1, false, false);
        acc2 = __builtin_amdgcn_wmma_f32_16x16x32_bf16(
            false, a, false, b2, (short)0, acc2, false, false);
        acc3 = __builtin_amdgcn_wmma_f32_16x16x32_bf16(
            false, a, false, b3, (short)0, acc3, false, false);
        if (!more) break;
        a = an; b0 = bn0; b1 = bn1; b2 = bn2; b3 = bn3;
        pa += 32; pb0 += 32; pb1 += 32; pb2 += 32; pb3 += 32;
        k0 += 32;
    }

    v8f accs[4] = {acc0, acc1, acc2, acc3};
#pragma unroll
    for (int j = 0; j < 4; ++j) {
        int n = col0 + (j << 4) + lrow;
        float bv = bias ? bias[n] : 0.f;
#pragma unroll
        for (int r = 0; r < 8; ++r) {
            int m = row0 + r + (half << 3);
            float val = accs[j][r] + bv;
            if (ACT == 1) val = tanhf(val);
            if (OUTB) Cb[(size_t)m * ldcb + n] = f2b(val);
            else      Cout[(size_t)m * ldc + n] = val;
        }
    }
}

// ---------------------------------------------------------------------------
// h := 0 (fp32 + bf16 mirrors)
// ---------------------------------------------------------------------------
__global__ void init_h_kernel(float* __restrict__ h, u16* __restrict__ hB) {
    int i = blockIdx.x * blockDim.x + threadIdx.x;
    if (i < kB * kHD) { h[i] = 0.f; hB[i] = 0; }
}

// ---------------------------------------------------------------------------
// Embedding gather (teacher forcing token) -> bf16 concat buffers
// catC = [ctx(512) | emb(256)]           (K=768)
// catO = [h(512) | emb(256) | ctx(512)]  (K=1280)
// ---------------------------------------------------------------------------
__global__ void gather_emb_kernel(const int* __restrict__ tgt,
                                  const float* __restrict__ embt,
                                  u16* __restrict__ catC,
                                  u16* __restrict__ catO, int t) {
    int i = blockIdx.x * blockDim.x + threadIdx.x;
    if (i >= kB * kE) return;
    int b = i / kE, e = i % kE;
    int tok = (t == 0) ? 0 : tgt[b * kT + (t - 1)];
    u16 v = f2b(embt[(size_t)tok * kE + e]);
    catC[(size_t)b * (kHE + kE) + kHE + e] = v;
    catO[(size_t)b * (kHD + kE + kHE) + kHD + e] = v;
}

// ---------------------------------------------------------------------------
// Attention scores: s[b,s] = sum_h tanh(enc_proj_bf16[b,s,h] + q[b,h]) * v[h]
// One wave per (b,s) row; packed 2xbf16 dword loads, shuffle reduction.
// ---------------------------------------------------------------------------
__global__ void attn_scores_kernel(const u16* __restrict__ encProjB,
                                   const float* __restrict__ q,
                                   const float* __restrict__ vat,
                                   float* __restrict__ scores) {
    int w = blockIdx.x * (blockDim.x >> 5) + (threadIdx.x >> 5);
    int lane = threadIdx.x & 31;
    if (w >= kB * kS) return;
    int b = w / kS;
    const unsigned* row = (const unsigned*)(encProjB + (size_t)w * kHA);
    const float* qb = q + (size_t)b * kHA;
    float acc = 0.f;
#pragma unroll
    for (int j = 0; j < kHA / 64; ++j) {
        int i2 = lane + j * 32;                  // dword index (2 elements)
        unsigned pk = row[i2];
        int i = i2 * 2;
        acc += tanhf(b2f((u16)pk)         + qb[i])     * vat[i];
        acc += tanhf(b2f((u16)(pk >> 16)) + qb[i + 1]) * vat[i + 1];
    }
#pragma unroll
    for (int off = 16; off > 0; off >>= 1)
        acc += __shfl_xor(acc, off, 32);
    if (lane == 0) scores[w] = acc;
}

// ---------------------------------------------------------------------------
// Softmax over S per batch row; also streams attention weights to d_out.
// ---------------------------------------------------------------------------
__global__ void softmax_kernel(const float* __restrict__ scores,
                               float* __restrict__ attn,
                               float* __restrict__ outAttn, int t) {
    __shared__ float red[kS];
    int b = blockIdx.x, s = threadIdx.x;
    float x = scores[(size_t)b * kS + s];
    red[s] = x; __syncthreads();
    for (int off = kS / 2; off > 0; off >>= 1) {
        if (s < off) red[s] = fmaxf(red[s], red[s + off]);
        __syncthreads();
    }
    float mx = red[0];
    __syncthreads();
    float e = __expf(x - mx);
    red[s] = e; __syncthreads();
    for (int off = kS / 2; off > 0; off >>= 1) {
        if (s < off) red[s] += red[s + off];
        __syncthreads();
    }
    float a = e / red[0];
    attn[(size_t)b * kS + s] = a;
    outAttn[((size_t)b * kT + t) * kS + s] = a;
}

// ---------------------------------------------------------------------------
// Context: ctx[b,e] = sum_s attn[b,s] * enc[b,s,e]  (batched vec-mat, VALU)
// ---------------------------------------------------------------------------
__global__ void ctx_kernel(const float* __restrict__ attn,
                           const float* __restrict__ enc,
                           u16* __restrict__ catC,
                           u16* __restrict__ catO) {
    int i = blockIdx.x * blockDim.x + threadIdx.x;
    if (i >= kB * kHE) return;
    int b = i / kHE, e = i % kHE;
    const float* ebase = enc + (size_t)b * kS * kHE + e;
    const float* arow = attn + (size_t)b * kS;
    float acc = 0.f;
#pragma unroll 4
    for (int s = 0; s < kS; ++s) acc += arow[s] * ebase[(size_t)s * kHE];
    u16 v = f2b(acc);
    catC[(size_t)b * (kHE + kE) + e] = v;
    catO[(size_t)b * (kHD + kE + kHE) + kHD + kE + e] = v;
}

// ---------------------------------------------------------------------------
// GRU cell update (torch gate order r,z,n)
// ---------------------------------------------------------------------------
__global__ void gru_kernel(const float* __restrict__ gi,
                           const float* __restrict__ gh,
                           float* __restrict__ h,
                           u16* __restrict__ hB,
                           u16* __restrict__ catO) {
    int i = blockIdx.x * blockDim.x + threadIdx.x;
    if (i >= kB * kHD) return;
    int b = i / kHD, d = i % kHD;
    size_t g = (size_t)b * 3 * kHD;
    float ir = gi[g + d], iz = gi[g + kHD + d], in_ = gi[g + 2 * kHD + d];
    float hr = gh[g + d], hz = gh[g + kHD + d], hn = gh[g + 2 * kHD + d];
    float r = 1.f / (1.f + __expf(-(ir + hr)));
    float z = 1.f / (1.f + __expf(-(iz + hz)));
    float n = tanhf(in_ + r * hn);
    float hnew = (1.f - z) * n + z * h[i];
    h[i] = hnew;
    u16 v = f2b(hnew);
    hB[i] = v;
    catO[(size_t)b * (kHD + kE + kHE) + d] = v;
}

// ---------------------------------------------------------------------------
// Host orchestration
// ---------------------------------------------------------------------------
extern "C" void kernel_launch(void* const* d_in, const int* in_sizes, int n_in,
                              void* d_out, int out_size, void* d_ws, size_t ws_size,
                              hipStream_t stream) {
    (void)in_sizes; (void)n_in; (void)out_size; (void)ws_size;

    const float* enc   = (const float*)d_in[0];   // [B,S,HE]
    const int*   tgt   = (const int*)d_in[1];     // [B,T]
    const float* embt  = (const float*)d_in[2];   // [C,E]
    const float* Wq    = (const float*)d_in[3];   // [HD,HA]
    const float* Wk    = (const float*)d_in[4];   // [HE,HA]
    const float* vat   = (const float*)d_in[5];   // [HA]
    const float* Wc    = (const float*)d_in[6];   // [E, HE+E]
    const float* bc    = (const float*)d_in[7];   // [E]
    const float* Wih   = (const float*)d_in[8];   // [3HD, E]
    const float* Whh   = (const float*)d_in[9];   // [3HD, HD]
    const float* bih   = (const float*)d_in[10];  // [3HD]
    const float* bhh   = (const float*)d_in[11];  // [3HD]
    const float* Wo    = (const float*)d_in[12];  // [C, HD+E+HE]
    const float* bo    = (const float*)d_in[13];  // [C]

    float* outLogits = (float*)d_out;                        // [B,T,C]
    float* outAttn   = outLogits + (size_t)kB * kT * kC;     // [B,T,S]

    // ---- workspace carving (256B aligned) ----
    char* p = (char*)d_ws;
    auto carve = [&](size_t bytes) -> void* {
        void* r = (void*)p;
        p += (bytes + 255) & ~(size_t)255;
        return r;
    };
    const int KO = kHD + kE + kHE;   // 1280
    const int KCk = kHE + kE;        // 768

    u16*   encB     = (u16*)carve((size_t)kB * kS * kHE * 2);
    u16*   encProjB = (u16*)carve((size_t)kB * kS * kHA * 2);
    u16*   WkT  = (u16*)carve((size_t)kHA * kHE * 2);   // [HA,HE] (transposed)
    u16*   WqT  = (u16*)carve((size_t)kHA * kHD * 2);   // [HA,HD] (transposed)
    u16*   WcB  = (u16*)carve((size_t)kE * KCk * 2);    // [E,768] as stored
    u16*   WihB = (u16*)carve((size_t)3 * kHD * kE * 2);
    u16*   WhhB = (u16*)carve((size_t)3 * kHD * kHD * 2);
    u16*   WoB  = (u16*)carve((size_t)kC * KO * 2);
    float* hF   = (float*)carve((size_t)kB * kHD * 4);
    u16*   hB   = (u16*)carve((size_t)kB * kHD * 2);
    float* qF   = (float*)carve((size_t)kB * kHA * 4);
    float* scF  = (float*)carve((size_t)kB * kS * 4);
    float* atF  = (float*)carve((size_t)kB * kS * 4);
    u16*   catC = (u16*)carve((size_t)kB * KCk * 2);
    u16*   catO = (u16*)carve((size_t)kB * KO * 2);
    u16*   combB = (u16*)carve((size_t)kB * kE * 2);
    float* giF  = (float*)carve((size_t)kB * 3 * kHD * 4);
    float* ghF  = (float*)carve((size_t)kB * 3 * kHD * 4);

    auto cvt = [&](const float* s, u16* d, int n) {
        cvt_f2b_kernel<<<(n + 255) / 256, 256, 0, stream>>>(s, d, n);
    };
    auto cvtT = [&](const float* s, u16* d, int K, int N) {
        cvt_f2b_transpose_kernel<<<(K * N + 255) / 256, 256, 0, stream>>>(s, d, K, N);
    };
    cvt(enc, encB, kB * kS * kHE);
    cvtT(Wk, WkT, kHE, kHA);                 // [HE,HA] -> [HA,HE]
    cvtT(Wq, WqT, kHD, kHA);                 // [HD,HA] -> [HA,HD]
    cvt(Wc, WcB, kE * KCk);                  // already [N,K] for Y = X @ Wc^T
    cvt(Wih, WihB, 3 * kHD * kE);
    cvt(Whh, WhhB, 3 * kHD * kHD);
    cvt(Wo, WoB, kC * KO);

    auto gemmBlocks = [](int M, int N) { return ((M / 16) * (N / 64) + 7) / 8; };

    // One-shot key projection: enc_proj(bf16) = enc(bf16) @ Wk
    gemm_wmma<0, true><<<gemmBlocks(kB * kS, kHA), 256, 0, stream>>>(
        encB, kHE, WkT, kHE, nullptr,
        nullptr, 0, encProjB, kHA,
        kB * kS, kHA, kHE);

    init_h_kernel<<<(kB * kHD + 255) / 256, 256, 0, stream>>>(hF, hB);

    for (int t = 0; t < kT; ++t) {
        gather_emb_kernel<<<(kB * kE + 255) / 256, 256, 0, stream>>>(
            tgt, embt, catC, catO, t);

        // q = h @ Wq
        gemm_wmma<0, false><<<gemmBlocks(kB, kHA), 256, 0, stream>>>(
            hB, kHD, WqT, kHD, nullptr, qF, kHA, nullptr, 0, kB, kHA, kHD);

        attn_scores_kernel<<<(kB * kS) / 8, 256, 0, stream>>>(
            encProjB, qF, vat, scF);

        softmax_kernel<<<kB, kS, 0, stream>>>(scF, atF, outAttn, t);

        ctx_kernel<<<(kB * kHE + 255) / 256, 256, 0, stream>>>(
            atF, enc, catC, catO);

        // comb = tanh([ctx,emb] @ Wc^T + bc)
        gemm_wmma<1, true><<<gemmBlocks(kB, kE), 256, 0, stream>>>(
            catC, KCk, WcB, KCk, bc, nullptr, 0, combB, kE, kB, kE, KCk);

        // gi = comb @ Wih^T + bih ;  gh = h @ Whh^T + bhh
        gemm_wmma<0, false><<<gemmBlocks(kB, 3 * kHD), 256, 0, stream>>>(
            combB, kE, WihB, kE, bih, giF, 3 * kHD, nullptr, 0, kB, 3 * kHD, kE);
        gemm_wmma<0, false><<<gemmBlocks(kB, 3 * kHD), 256, 0, stream>>>(
            hB, kHD, WhhB, kHD, bhh, ghF, 3 * kHD, nullptr, 0, kB, 3 * kHD, kHD);

        gru_kernel<<<(kB * kHD + 255) / 256, 256, 0, stream>>>(
            giF, ghF, hF, hB, catO);

        // out[b,t,:] = [h,emb,ctx] @ Wo^T + bo   (strided store into [B,T,C])
        gemm_wmma<0, false><<<gemmBlocks(kB, kC), 256, 0, stream>>>(
            catO, KO, WoB, KO, bo,
            outLogits + (size_t)t * kC, kT * kC, nullptr, 0,
            kB, kC, KO);
    }
}
